// Autoformer_40278203302269
// MI455X (gfx1250) — matrix operand order repaired and checked by reading, not compiled, not run on hardware
//
#include <hip/hip_runtime.h>

typedef __attribute__((ext_vector_type(16))) __bf16 bf16x16;
typedef __attribute__((ext_vector_type(8)))  float  f32x8;
typedef __attribute__((ext_vector_type(4)))  unsigned int u32x4;
typedef __attribute__((ext_vector_type(8)))  int i32x8;
typedef __attribute__((ext_vector_type(4)))  int i32x4;

#define TOPK 31
#define KC 128            // K-chunk staged per TDM transfer (bf16 elements)

__device__ __forceinline__ unsigned short af_f2bf(float f) {
  unsigned int u = __float_as_uint(f);
  u += 0x7FFFu + ((u >> 16) & 1u);          // round-to-nearest-even
  return (unsigned short)(u >> 16);
}
__device__ __forceinline__ float af_gelu(float x) {
  return 0.5f * x * (1.0f + erff(x * 0.7071067811865475f));
}

// Issue one TDM 2D tile load: 64 rows x KC cols of bf16 from A[row0 + r][k0 + c]
// (row stride K elements) into LDS at byte offset lds_off. Rows beyond M read 0.
__device__ __forceinline__ void af_tdm_issue(const unsigned short* A, int row0, int k0,
                                             int M, int K, unsigned int lds_off) {
  unsigned long long gaddr =
      (unsigned long long)(const void*)A + ((size_t)row0 * K + k0) * 2ull;
  int remM = M - row0;                       // remaining rows from tile start
  unsigned int remK = (unsigned int)(K - k0);
  u32x4 g0; i32x8 g1;
  i32x4 g2 = {0, 0, 0, 0}, g3 = {0, 0, 0, 0};
  i32x8 g4 = {0, 0, 0, 0, 0, 0, 0, 0};
  g0[0] = 1u;                                               // count=1 user descriptor
  g0[1] = lds_off;                                          // lds_addr (bytes)
  g0[2] = (unsigned int)gaddr;                              // global_addr[31:0]
  g0[3] = (unsigned int)((gaddr >> 32) & 0x1FFFFFFull) | (2u << 30);  // addr hi | type=2
  g1[0] = (int)(1u << 16);                                  // data_size=1 (2 bytes)
  g1[1] = (int)((remK & 0xFFFFu) << 16);                    // tensor_dim0[15:0] @bit48
  g1[2] = (int)((remK >> 16) | ((unsigned)(remM & 0xFFFF) << 16)); // dim0 hi | dim1 lo
  g1[3] = (int)((((unsigned)remM >> 16) & 0xFFFFu) | ((unsigned)KC << 16)); // dim1 hi | tile_dim0
  g1[4] = 64;                                               // tile_dim1=64, tile_dim2=0
  g1[5] = (int)(unsigned)K;                                 // tensor_dim0_stride lo32
  g1[6] = 0;                                                // stride hi | dim1_stride lo
  g1[7] = 0;
  __builtin_amdgcn_tensor_load_to_lds(g0, g1, g2, g3, g4, 0);
}

// ---------------------------------------------------------------------------
// bf16 WMMA GEMM:  C[M,N] = act( A[M,K] * W[K,N] + bias )
// A: bf16 row-major (M x K).  Wt: bf16 "N-major" (N x K) == W transposed.
// Block = 8 waves (2x4) computes a 64x256 C tile; each wave 32x64 = 2x4 wmma
// tiles (8 independent accumulators -> no WMMA hazard nops).
// A tile (64 x KC) staged into LDS by the Tensor Data Mover, double-buffered
// on TENSORcnt so DMA of chunk c+1 overlaps WMMA on chunk c. B streams from
// global (per-wave-unique columns, contiguous along K -> b128 clauses).
// ---------------------------------------------------------------------------
__global__ __launch_bounds__(256) void af_gemm(
    const unsigned short* __restrict__ A, long long sAb,
    const unsigned short* __restrict__ Wt, long long sWb,
    const float* __restrict__ bias,
    float* __restrict__ Cout, long long sCb,
    int M, int N, int K, int act)
{
  __shared__ unsigned short Abuf[2][64 * KC];   // 2 x 16 KB, LDS offset 0

  // Escape the LDS buffer so the compiler must treat it as written by the TDM
  // (the tensor_load_to_lds builtin carries no pointer-level data dependence).
  {
    unsigned short* esc = &Abuf[0][0];
    asm volatile("" : "+v"(esc) : : "memory");
  }

  const int bz = blockIdx.y;
  A    += (size_t)bz * sAb;
  Wt   += (size_t)bz * sWb;
  Cout += (size_t)bz * sCb;

  const int NB   = N >> 8;                  // 256-wide column blocks
  const int nb   = blockIdx.x % NB;
  const int mb   = blockIdx.x / NB;
  const int row0 = mb * 64;
  const int col0 = nb * 256;

  const int wave  = threadIdx.x >> 5;
  const int lane  = threadIdx.x & 31;
  const int half  = lane >> 4;
  const int lan16 = lane & 15;
  const int wm = (wave >> 2) * 32;          // wave row offset inside block tile
  const int wn = (wave & 3) * 64;           // wave col offset inside block tile
  const int uwave = __builtin_amdgcn_readfirstlane(wave);  // uniform wave id

  const unsigned int* Brow[4];
#pragma unroll
  for (int ni = 0; ni < 4; ++ni)
    Brow[ni] = (const unsigned int*)(Wt + (size_t)(col0 + wn + ni * 16 + lan16) * K);

  f32x8 acc[2][4] = {};
  union UA { bf16x16 v; unsigned int u[8]; } afrag[2], bfrag[4];

  const int NCH = K / KC;
  if (uwave == 0) af_tdm_issue(A, row0, 0, M, K, 0u);  // prefetch chunk 0

  for (int c = 0; c < NCH; ++c) {
    if (uwave == 0) {
      if (c + 1 < NCH) {
        af_tdm_issue(A, row0, (c + 1) * KC, M, K, (unsigned)(((c + 1) & 1) * 64 * KC * 2));
        __builtin_amdgcn_s_wait_tensorcnt(1);   // chunk c has landed
      } else {
        __builtin_amdgcn_s_wait_tensorcnt(0);
      }
    }
    __syncthreads();                            // chunk c visible to all waves
    asm volatile("" ::: "memory");              // order LDS reads after the DMA

    const unsigned int* Al0 = (const unsigned int*)&Abuf[c & 1][0] + (((wm + lan16) * KC) >> 1);
    const unsigned int* Al1 = (const unsigned int*)&Abuf[c & 1][0] + (((wm + 16 + lan16) * KC) >> 1);
    const int kbase = c * KC;
#pragma unroll
    for (int kin = 0; kin < KC; kin += 32) {
#pragma unroll
      for (int j = 0; j < 8; ++j) {
        // A 16x32 bf16 layout: vgpr j holds K pair at (j>>2)*16 + half*8 + (j&3)*2
        const int aoff = (j >> 2) * 8 + half * 4 + (j & 3) + (kin >> 1);
        afrag[0].u[j] = Al0[aoff];
        afrag[1].u[j] = Al1[aoff];
        // B 32x16 bf16 layout: vgpr j holds K pair at (j>>2)*16 + half*8 + (j&3)*2  (N = lane&15)
        const int boff = ((kbase + kin) >> 1) + (j >> 2) * 8 + half * 4 + (j & 3);
        bfrag[0].u[j] = Brow[0][boff];
        bfrag[1].u[j] = Brow[1][boff];
        bfrag[2].u[j] = Brow[2][boff];
        bfrag[3].u[j] = Brow[3][boff];
      }
#pragma unroll
      for (int mi = 0; mi < 2; ++mi)
#pragma unroll
        for (int ni = 0; ni < 4; ++ni)
          acc[mi][ni] = __builtin_amdgcn_wmma_f32_16x16x32_bf16(
              false, afrag[mi].v, false, bfrag[ni].v,
              (short)0, acc[mi][ni], false, false);
    }
    __syncthreads();                            // done with buf[c&1] before re-fill
  }

  // epilogue: bias + optional exact GELU, guarded stores for ragged M
#pragma unroll
  for (int ni = 0; ni < 4; ++ni) {
    const int ncol = col0 + wn + ni * 16 + lan16;
    const float bv = bias ? bias[ncol] : 0.0f;
#pragma unroll
    for (int mi = 0; mi < 2; ++mi) {
#pragma unroll
      for (int r = 0; r < 8; ++r) {
        const int m = row0 + wm + mi * 16 + half * 8 + r;  // vgpr r -> row r (+8 hi half)
        if (m < M) {
          float v = acc[mi][ni][r] + bv;
          if (act == 1) v = af_gelu(v);
          Cout[(size_t)m * N + ncol] = v;
        }
      }
    }
  }
}

// --------------------------- pointwise / small kernels ---------------------
__global__ void af_wpack(const float* __restrict__ w, unsigned short* __restrict__ wt,
                         int K, int N) {  // w: K x N row-major -> wt: N x K bf16
  size_t i = (size_t)blockIdx.x * blockDim.x + threadIdx.x;
  if (i >= (size_t)K * N) return;
  int n = (int)(i / K), k = (int)(i % K);
  wt[i] = af_f2bf(w[(size_t)k * N + n]);
}

__global__ void af_cvt(const float* __restrict__ x, unsigned short* __restrict__ y,
                       size_t n) {
  size_t i = (size_t)blockIdx.x * blockDim.x + threadIdx.x;
  if (i < n) y[i] = af_f2bf(x[i]);
}

__global__ void af_add(const float* __restrict__ a, const float* __restrict__ b,
                       float* __restrict__ o, size_t n) {
  size_t i = (size_t)blockIdx.x * blockDim.x + threadIdx.x;
  if (i < n) o[i] = a[i] + b[i];
}

__global__ void af_xt(const float* __restrict__ xe, unsigned short* __restrict__ xt,
                      int B, int L, int C) {  // (B,L,C) -> bf16 (B*C, L)
  size_t i = (size_t)blockIdx.x * blockDim.x + threadIdx.x;
  if (i >= (size_t)B * C * L) return;
  int l = (int)(i % L), c = (int)((i / L) % C), b = (int)(i / ((size_t)C * L));
  xt[i] = af_f2bf(xe[((size_t)b * L + l) * C + c]);
}

__global__ void af_xuntr(const float* __restrict__ xt2, float* __restrict__ x,
                         int B, int L, int C) {  // (B*C, L) -> (B,L,C)
  size_t i = (size_t)blockIdx.x * blockDim.x + threadIdx.x;
  if (i >= (size_t)B * L * C) return;
  int c = (int)(i % C), l = (int)((i / C) % L), b = (int)(i / ((size_t)L * C));
  x[i] = xt2[((size_t)b * C + c) * L + l];
}

__global__ void af_decomp(const float* __restrict__ x, float* seas, float* trend,
                          int B, int L, int C) {  // 25-tap replicate-pad moving avg
  size_t i = (size_t)blockIdx.x * blockDim.x + threadIdx.x;
  if (i >= (size_t)B * L * C) return;
  int c = (int)(i % C), l = (int)((i / C) % L), b = (int)(i / ((size_t)L * C));
  const float* xb = x + (size_t)b * L * C + c;
  float s = 0.f;
#pragma unroll
  for (int j = -12; j <= 12; ++j) {
    int ll = l + j; ll = ll < 0 ? 0 : (ll > L - 1 ? L - 1 : ll);
    s += xb[(size_t)ll * C];
  }
  s *= (1.0f / 25.0f);
  if (trend) trend[i] = s;
  if (seas)  seas[i]  = xb[(size_t)l * C] - s;
}

__global__ void af_meanL(const float* __restrict__ x, float* __restrict__ mean,
                         int B, int L, int C) {
  int i = blockIdx.x * blockDim.x + threadIdx.x;
  if (i >= B * C) return;
  int b = i / C, c = i % C;
  float s = 0.f;
  for (int l = 0; l < L; ++l) s += x[((size_t)b * L + l) * C + c];
  mean[i] = s / (float)L;
}

__global__ void af_inits(const float* __restrict__ seas, const float* __restrict__ trend,
                         const float* __restrict__ mean, float* __restrict__ seas_init,
                         float* __restrict__ trend_acc, int B, int L, int C) {
  size_t i = (size_t)blockIdx.x * blockDim.x + threadIdx.x;
  if (i >= (size_t)B * 512 * C) return;
  int c = (int)(i % C), t = (int)((i / C) % 512), b = (int)(i / ((size_t)512 * C));
  if (t < 256) {
    size_t s = ((size_t)b * L + 256 + t) * C + c;   // rows L-256..L-1
    seas_init[i] = seas[s];
    trend_acc[i] = trend[s];
  } else {
    seas_init[i] = 0.f;
    trend_acc[i] = mean[b * C + c];
  }
}

__global__ void af_embed(const float* __restrict__ src, int C,
                         const float* __restrict__ convw, const float* __restrict__ mark,
                         const float* __restrict__ timew, float* __restrict__ out,
                         int B, int Ls, int D) {
  size_t i = (size_t)blockIdx.x * blockDim.x + threadIdx.x;
  if (i >= (size_t)B * Ls * D) return;
  int d = (int)(i % D), l = (int)((i / D) % Ls), b = (int)(i / ((size_t)Ls * D));
  int lm = (l + Ls - 1) % Ls, lp = (l + 1) % Ls;
  const float* r0 = src + ((size_t)b * Ls + lm) * C;
  const float* r1 = src + ((size_t)b * Ls + l ) * C;
  const float* r2 = src + ((size_t)b * Ls + lp) * C;
  float acc = 0.f;
  for (int c = 0; c < C; ++c)
    acc += r0[c] * convw[(0 * C + c) * D + d] +
           r1[c] * convw[(1 * C + c) * D + d] +
           r2[c] * convw[(2 * C + c) * D + d];
  const float* mk = mark + ((size_t)b * Ls + l) * 4;
#pragma unroll
  for (int m = 0; m < 4; ++m) acc += mk[m] * timew[m * D + d];
  int i2 = d >> 1;
  float ang = (float)l * expf(-(float)(2 * i2) * 9.210340371976184f / (float)D);
  acc += (d & 1) ? cosf(ang) : sinf(ang);
  out[i] = acc;
}

__global__ __launch_bounds__(256) void af_diag(const float* __restrict__ S,
                                               float* __restrict__ corr, int B, int L) {
  __shared__ float red[256];
  int b = blockIdx.x / L, tau = blockIdx.x % L;
  const float* Sb = S + (size_t)b * L * L;
  float s = 0.f;
  for (int t = threadIdx.x; t < L; t += 256)
    s += Sb[(size_t)((t + tau) & (L - 1)) * L + t];
  red[threadIdx.x] = s; __syncthreads();
  for (int st = 128; st > 0; st >>= 1) {
    if ((int)threadIdx.x < st) red[threadIdx.x] += red[threadIdx.x + st];
    __syncthreads();
  }
  if (threadIdx.x == 0) corr[b * L + tau] = red[0] * (1.0f / 512.0f);  // mean over H*E
}

__global__ __launch_bounds__(256) void af_topk(const float* __restrict__ corr,
                                               float* __restrict__ w,
                                               int* __restrict__ delay, int L) {
  __shared__ float vals[512];
  __shared__ float rv[256];
  __shared__ int   ri[256];
  __shared__ float tw[TOPK];
  int b = blockIdx.x;
  for (int t = threadIdx.x; t < L; t += 256) vals[t] = corr[b * L + t];
  __syncthreads();
  for (int kk = 0; kk < TOPK; ++kk) {
    float bv = -3.0e38f; int bi = 0;
    for (int t = threadIdx.x; t < L; t += 256) {
      float v = vals[t];
      if (v > bv) { bv = v; bi = t; }
    }
    rv[threadIdx.x] = bv; ri[threadIdx.x] = bi;
    __syncthreads();
    for (int st = 128; st > 0; st >>= 1) {
      if ((int)threadIdx.x < st) {
        if (rv[threadIdx.x + st] > rv[threadIdx.x] ||
            (rv[threadIdx.x + st] == rv[threadIdx.x] &&
             ri[threadIdx.x + st] < ri[threadIdx.x])) {
          rv[threadIdx.x] = rv[threadIdx.x + st];
          ri[threadIdx.x] = ri[threadIdx.x + st];
        }
      }
      __syncthreads();
    }
    if (threadIdx.x == 0) { tw[kk] = rv[0]; delay[b * TOPK + kk] = ri[0]; vals[ri[0]] = -3.0e38f; }
    __syncthreads();
  }
  if (threadIdx.x == 0) {
    float mx = tw[0], ssum = 0.f, ex[TOPK];
    for (int kk = 0; kk < TOPK; ++kk) { ex[kk] = expf(tw[kk] - mx); ssum += ex[kk]; }
    for (int kk = 0; kk < TOPK; ++kk) w[b * TOPK + kk] = ex[kk] / ssum;
  }
}

__global__ void af_gather(const float* __restrict__ v, const float* __restrict__ w,
                          const int* __restrict__ delay, float* __restrict__ out,
                          int B, int L, int D) {
  size_t i = (size_t)blockIdx.x * blockDim.x + threadIdx.x;
  if (i >= (size_t)B * L * D) return;
  int d = (int)(i % D), l = (int)((i / D) % L), b = (int)(i / ((size_t)L * D));
  const float* wb = w + b * TOPK; const int* db = delay + b * TOPK;
  float acc = 0.f;
  for (int kk = 0; kk < TOPK; ++kk) {
    int ll = (l + db[kk]) & (L - 1);
    acc += wb[kk] * v[((size_t)b * L + ll) * D + d];
  }
  out[i] = acc;
}

__global__ __launch_bounds__(256) void af_ln(const float* x, const float* __restrict__ g,
                                             const float* __restrict__ bta, float* y, int D) {
  __shared__ float r1[256], r2[256];
  size_t row = blockIdx.x;
  const float* xr = x + row * D;
  float s = 0.f, sq = 0.f;
  for (int d = threadIdx.x; d < D; d += 256) { float v = xr[d]; s += v; sq += v * v; }
  r1[threadIdx.x] = s; r2[threadIdx.x] = sq; __syncthreads();
  for (int st = 128; st > 0; st >>= 1) {
    if ((int)threadIdx.x < st) { r1[threadIdx.x] += r1[threadIdx.x + st];
                                 r2[threadIdx.x] += r2[threadIdx.x + st]; }
    __syncthreads();
  }
  float mu = r1[0] / D, var = r2[0] / D - mu * mu, rs = rsqrtf(var + 1e-5f);
  for (int d = threadIdx.x; d < D; d += 256)
    y[row * D + d] = (xr[d] - mu) * rs * g[d] + bta[d];
}

__global__ void af_centerL(float* x, int B, int L, int D) {
  int i = blockIdx.x * blockDim.x + threadIdx.x;
  if (i >= B * D) return;
  int b = i / D, d = i % D;
  float* xb = x + (size_t)b * L * D + d;
  float s = 0.f;
  for (int l = 0; l < L; ++l) s += xb[(size_t)l * D];
  s /= (float)L;
  for (int l = 0; l < L; ++l) xb[(size_t)l * D] -= s;
}

__global__ void af_trendconv(const float* __restrict__ t, const float* __restrict__ w,
                             float* __restrict__ acc, int B, int L, int D, int C) {
  size_t i = (size_t)blockIdx.x * blockDim.x + threadIdx.x;
  if (i >= (size_t)B * L * C) return;
  int c = (int)(i % C), l = (int)((i / C) % L), b = (int)(i / ((size_t)L * C));
  int lm = (l + L - 1) % L, lp = (l + 1) % L;
  const float* r0 = t + ((size_t)b * L + lm) * D;
  const float* r1 = t + ((size_t)b * L + l ) * D;
  const float* r2 = t + ((size_t)b * L + lp) * D;
  float a = 0.f;
  for (int d = 0; d < D; ++d)
    a += r0[d] * w[(0 * D + d) * C + c] + r1[d] * w[(1 * D + d) * C + c] +
         r2[d] * w[(2 * D + d) * C + c];
  acc[i] += a;
}

__global__ void af_final(const float* __restrict__ xs, const float* __restrict__ pw,
                         const float* __restrict__ pb, const float* __restrict__ tacc,
                         float* __restrict__ out, int B, int L, int D, int C, int P) {
  size_t i = (size_t)blockIdx.x * blockDim.x + threadIdx.x;
  if (i >= (size_t)B * P * C) return;
  int c = (int)(i % C), t = (int)((i / C) % P), b = (int)(i / ((size_t)P * C));
  int l = L - P + t;
  const float* xr = xs + ((size_t)b * L + l) * D;
  float a = pb[c];
  for (int d = 0; d < D; ++d) a += xr[d] * pw[d * C + c];
  out[i] = a + tacc[((size_t)b * L + l) * C + c];
}

// --------------------------------- host ------------------------------------
static inline unsigned gdiv(size_t n) { return (unsigned)((n + 255) >> 8); }

extern "C" void kernel_launch(void* const* d_in, const int* in_sizes, int n_in,
                              void* d_out, int out_size, void* d_ws, size_t ws_size,
                              hipStream_t stream) {
  (void)in_sizes; (void)n_in; (void)out_size; (void)ws_size;
  constexpr int B = 8, L = 512, C = 21, D = 512, F = 2048, EL = 3, DL = 2, PRED = 256, Ld = 512;
  const size_t BLD = (size_t)B * L * D, BLC = (size_t)B * L * C;
  const size_t BLF = (size_t)B * L * F, BLL = (size_t)B * L * L;

  const float* x_enc      = (const float*)d_in[0];
  const float* x_mark_enc = (const float*)d_in[1];
  const float* x_mark_dec = (const float*)d_in[3];
  const float* space_w1   = (const float*)d_in[4];
  const float* space_b1   = (const float*)d_in[5];
  const float* space_w2   = (const float*)d_in[6];
  const float* space_b2   = (const float*)d_in[7];
  const float* enc_conv_w = (const float*)d_in[8];
  const float* enc_time_w = (const float*)d_in[9];
  const float* dec_conv_w = (const float*)d_in[10];
  const float* dec_time_w = (const float*)d_in[11];
  const float *enc_W[4], *enc_b[4], *dec_sW[4], *dec_sb[4], *dec_cW[4], *dec_cb[4];
  for (int p = 0; p < 4; ++p) {          // setup_inputs dict order: q,k,v,o interleaved
    enc_W[p]  = (const float*)d_in[12 + p * 6 + 0];
    enc_b[p]  = (const float*)d_in[12 + p * 6 + 1];
    dec_sW[p] = (const float*)d_in[12 + p * 6 + 2];
    dec_sb[p] = (const float*)d_in[12 + p * 6 + 3];
    dec_cW[p] = (const float*)d_in[12 + p * 6 + 4];
    dec_cb[p] = (const float*)d_in[12 + p * 6 + 5];
  }
  const float* enc_ff1     = (const float*)d_in[36];
  const float* enc_ff2     = (const float*)d_in[37];
  const float* enc_ln_g    = (const float*)d_in[38];
  const float* enc_ln_b    = (const float*)d_in[39];
  const float* dec_ff1     = (const float*)d_in[40];
  const float* dec_ff2     = (const float*)d_in[41];
  const float* dec_trend_w = (const float*)d_in[42];
  const float* dec_ln_g    = (const float*)d_in[43];
  const float* dec_ln_b    = (const float*)d_in[44];
  const float* proj_w      = (const float*)d_in[45];
  const float* proj_b      = (const float*)d_in[46];

  // ---- bump allocator over workspace ----
  char* base = (char*)d_ws; size_t used = 0;
  auto alloc = [&](size_t bytes) -> void* {
    size_t o = (used + 255) & ~(size_t)255; used = o + bytes; return base + o;
  };
  auto fbuf = [&](size_t n) { return (float*)alloc(n * 4); };
  auto hbuf = [&](size_t n) { return (unsigned short*)alloc(n * 2); };

  // packed bf16 weights (N-major)
  unsigned short* sw1T = hbuf((size_t)512 * 512);
  unsigned short* sw2T = hbuf((size_t)512 * 512);
  unsigned short *encWT[4][EL], *decsWT[4][DL], *deccWT[4][DL];
  for (int p = 0; p < 4; ++p) {
    for (int l = 0; l < EL; ++l) encWT[p][l]  = hbuf((size_t)D * D);
    for (int l = 0; l < DL; ++l) decsWT[p][l] = hbuf((size_t)D * D);
    for (int l = 0; l < DL; ++l) deccWT[p][l] = hbuf((size_t)D * D);
  }
  unsigned short *eff1T[EL], *eff2T[EL], *dff1T[DL], *dff2T[DL];
  for (int l = 0; l < EL; ++l) { eff1T[l] = hbuf((size_t)D * F); eff2T[l] = hbuf((size_t)F * D); }
  for (int l = 0; l < DL; ++l) { dff1T[l] = hbuf((size_t)D * F); dff2T[l] = hbuf((size_t)F * D); }

  // activations
  unsigned short* xtbf = hbuf((size_t)B * C * L);
  float* h1   = fbuf((size_t)B * C * D);
  unsigned short* h1bf = hbuf((size_t)B * C * D);
  float* xt2  = fbuf((size_t)B * C * L);
  float* xbuf = fbuf(BLC);
  float* seasC = fbuf(BLC); float* trendC = fbuf(BLC); float* meanC = fbuf((size_t)B * C);
  float* seas_init = fbuf(BLC); float* trend_acc = fbuf(BLC);
  float* enc_out = fbuf(BLD); float* xs = fbuf(BLD);
  unsigned short* bfa = hbuf(BLD); unsigned short* bfb = hbuf(BLD);
  unsigned short* qbf = hbuf(BLD); unsigned short* kbf = hbuf(BLD);
  float* qf = fbuf(BLD); float* kf = fbuf(BLD); float* vf = fbuf(BLD);
  float* S  = fbuf(BLL);
  float* corrb = fbuf((size_t)B * L);
  float* wtop  = fbuf((size_t)B * TOPK);
  int*   dtop  = (int*)alloc((size_t)B * TOPK * 4);
  float* abuf = fbuf(BLD); float* resb = fbuf(BLD); float* ybuf = fbuf(BLD);
  float* t1 = fbuf(BLD); float* ttmp = fbuf(BLD);
  float* ffh = fbuf(BLF);
  unsigned short* ffbf = hbuf(BLF);

  // ---- helpers ----
  auto pack = [&](const float* src, unsigned short* dst, int K, int N) {
    size_t n = (size_t)K * N;
    af_wpack<<<gdiv(n), 256, 0, stream>>>(src, dst, K, N);
  };
  auto cvt = [&](const float* src, unsigned short* dst, size_t n) {
    af_cvt<<<gdiv(n), 256, 0, stream>>>(src, dst, n);
  };
  auto add = [&](const float* a, const float* b, float* o, size_t n) {
    af_add<<<gdiv(n), 256, 0, stream>>>(a, b, o, n);
  };
  auto gemm = [&](const unsigned short* A, long long sAb, const unsigned short* Wt,
                  long long sWb, const float* bias, float* Cm, long long sCb,
                  int M, int N, int K, int act, int batch) {
    int MB = (M + 63) / 64, NB = N / 256;
    dim3 grid((unsigned)(MB * NB), (unsigned)batch);
    af_gemm<<<grid, 256, 0, stream>>>(A, sAb, Wt, sWb, bias, Cm, sCb, M, N, K, act);
  };
  auto decomp = [&](const float* x, float* seas, float* trend, int Lx, int Cx) {
    size_t n = (size_t)B * Lx * Cx;
    af_decomp<<<gdiv(n), 256, 0, stream>>>(x, seas, trend, B, Lx, Cx);
  };
  auto corr_layer = [&](const float* xq, const float* xkv,
                        const unsigned short* WqT, const float* bq,
                        const unsigned short* WkT, const float* bk,
                        const unsigned short* WvT, const float* bv,
                        const unsigned short* WoT, const float* bo, float* aout) {
    cvt(xq, bfa, BLD);
    cvt(xkv, bfb, BLD);
    gemm(bfa, 0, WqT, 0, bq, qf, 0, B * L, D, D, 0, 1);
    gemm(bfb, 0, WkT, 0, bk, kf, 0, B * L, D, D, 0, 1);
    gemm(bfb, 0, WvT, 0, bv, vf, 0, B * L, D, D, 0, 1);
    cvt(qf, qbf, BLD);
    cvt(kf, kbf, BLD);
    // S[b] = Q[b] * K[b]^T : K[b] row-major (L,D) is already the N-major operand
    gemm(qbf, (long long)L * D, kbf, (long long)L * D, nullptr, S, (long long)L * L,
         L, L, D, 0, B);
    af_diag<<<B * L, 256, 0, stream>>>(S, corrb, B, L);
    af_topk<<<B, 256, 0, stream>>>(corrb, wtop, dtop, L);
    af_gather<<<gdiv(BLD), 256, 0, stream>>>(vf, wtop, dtop, qf, B, L, D); // reuse qf
    cvt(qf, bfa, BLD);
    gemm(bfa, 0, WoT, 0, bo, aout, 0, B * L, D, D, 0, 1);
  };
  auto ffn = [&](const float* xin, const unsigned short* f1T, const unsigned short* f2T,
                 float* yout) {
    cvt(xin, bfa, BLD);
    gemm(bfa, 0, f1T, 0, nullptr, ffh, 0, B * L, F, D, 1 /*gelu*/, 1);
    cvt(ffh, ffbf, BLF);
    gemm(ffbf, 0, f2T, 0, nullptr, yout, 0, B * L, D, F, 0, 1);
  };

  // ---- pack all weights to bf16 N-major ----
  pack(space_w1, sw1T, 512, 512);
  pack(space_w2, sw2T, 512, 512);
  for (int p = 0; p < 4; ++p) {
    for (int l = 0; l < EL; ++l) pack(enc_W[p]  + (size_t)l * D * D, encWT[p][l],  D, D);
    for (int l = 0; l < DL; ++l) pack(dec_sW[p] + (size_t)l * D * D, decsWT[p][l], D, D);
    for (int l = 0; l < DL; ++l) pack(dec_cW[p] + (size_t)l * D * D, deccWT[p][l], D, D);
  }
  for (int l = 0; l < EL; ++l) {
    pack(enc_ff1 + (size_t)l * D * F, eff1T[l], D, F);
    pack(enc_ff2 + (size_t)l * F * D, eff2T[l], F, D);
  }
  for (int l = 0; l < DL; ++l) {
    pack(dec_ff1 + (size_t)l * D * F, dff1T[l], D, F);
    pack(dec_ff2 + (size_t)l * F * D, dff2T[l], F, D);
  }

  // ---- space MLP over the channel axis ----
  af_xt<<<gdiv((size_t)B * C * L), 256, 0, stream>>>(x_enc, xtbf, B, L, C);
  gemm(xtbf, 0, sw1T, 0, space_b1, h1, 0, B * C, D, L, 1 /*gelu*/, 1);
  cvt(h1, h1bf, (size_t)B * C * D);
  gemm(h1bf, 0, sw2T, 0, space_b2, xt2, 0, B * C, L, D, 0, 1);
  af_xuntr<<<gdiv(BLC), 256, 0, stream>>>(xt2, xbuf, B, L, C);

  // ---- decomposition + decoder inits ----
  decomp(xbuf, seasC, trendC, L, C);
  af_meanL<<<1, 256, 0, stream>>>(xbuf, meanC, B, L, C);
  af_inits<<<gdiv(BLC), 256, 0, stream>>>(seasC, trendC, meanC, seas_init, trend_acc, B, L, C);

  // ---- embeddings ----
  af_embed<<<gdiv(BLD), 256, 0, stream>>>(xbuf, C, enc_conv_w, x_mark_enc, enc_time_w,
                                          enc_out, B, L, D);
  af_embed<<<gdiv(BLD), 256, 0, stream>>>(seas_init, C, dec_conv_w, x_mark_dec, dec_time_w,
                                          xs, B, Ld, D);

  // ---- encoder ----
  for (int l = 0; l < EL; ++l) {
    corr_layer(enc_out, enc_out,
               encWT[0][l], enc_b[0] + l * D, encWT[1][l], enc_b[1] + l * D,
               encWT[2][l], enc_b[2] + l * D, encWT[3][l], enc_b[3] + l * D, abuf);
    add(enc_out, abuf, resb, BLD);
    decomp(resb, abuf, nullptr, L, D);     // xx (seasonal only)
    ffn(abuf, eff1T[l], eff2T[l], ybuf);
    add(abuf, ybuf, resb, BLD);
    decomp(resb, enc_out, nullptr, L, D);
  }
  af_ln<<<B * L, 256, 0, stream>>>(enc_out, enc_ln_g, enc_ln_b, enc_out, D);
  af_centerL<<<gdiv((size_t)B * D), 256, 0, stream>>>(enc_out, B, L, D);

  // ---- decoder ----
  for (int l = 0; l < DL; ++l) {
    corr_layer(xs, xs,
               decsWT[0][l], dec_sb[0] + l * D, decsWT[1][l], dec_sb[1] + l * D,
               decsWT[2][l], dec_sb[2] + l * D, decsWT[3][l], dec_sb[3] + l * D, abuf);
    add(xs, abuf, resb, BLD);
    decomp(resb, xs, t1, Ld, D);
    corr_layer(xs, enc_out,
               deccWT[0][l], dec_cb[0] + l * D, deccWT[1][l], dec_cb[1] + l * D,
               deccWT[2][l], dec_cb[2] + l * D, deccWT[3][l], dec_cb[3] + l * D, abuf);
    add(xs, abuf, resb, BLD);
    decomp(resb, xs, ttmp, Ld, D);
    add(t1, ttmp, t1, BLD);
    ffn(xs, dff1T[l], dff2T[l], ybuf);
    add(xs, ybuf, resb, BLD);
    decomp(resb, xs, ttmp, Ld, D);
    add(t1, ttmp, t1, BLD);
    af_trendconv<<<gdiv(BLC), 256, 0, stream>>>(t1, dec_trend_w + (size_t)l * 3 * D * C,
                                                trend_acc, B, Ld, D, C);
  }
  af_ln<<<B * Ld, 256, 0, stream>>>(xs, dec_ln_g, dec_ln_b, xs, D);
  af_centerL<<<gdiv((size_t)B * D), 256, 0, stream>>>(xs, B, Ld, D);

  // ---- final projection + trend + slice ----
  af_final<<<gdiv((size_t)B * PRED * C), 256, 0, stream>>>(
      xs, proj_w, proj_b, trend_acc, (float*)d_out, B, Ld, D, C, PRED);
}
